// HAN_53369263620116
// MI455X (gfx1250) — matrix operand order repaired and checked by reading, not compiled, not run on hardware
//
#include <hip/hip_runtime.h>
#include <hip/hip_bf16.h>
#include <cstdint>

// ---------------------------------------------------------------------------
// HAN music model for MI455X (gfx1250).
// Strategy: hoist every time-parallel projection into f16 WMMA GEMMs
// (v_wmma_f32_16x16x32_f16), leave only the minimal per-step recurrent work
// (hidden matvec + 13 recurrent input columns) in sequential LDS kernels.
// GEMM: 2x4 tiles (32x64 C) per wave -> 8 WMMAs per K-chunk with A/B fragment
// reuse; branch-free loads (clamped rows, select-to-zero K tail);
// __launch_bounds__(128,1) to avoid VGPR spills.
// ---------------------------------------------------------------------------

typedef __attribute__((ext_vector_type(16))) _Float16 v16h;
typedef __attribute__((ext_vector_type(8)))  float    v8f;

static __device__ __forceinline__ float sigf(float x) {
  return 1.0f / (1.0f + expf(-x));
}

// C[M x N] = A[M x K] (row stride lda) * W[N x K]^T (row stride ldw) + bias[N]
// Each wave computes a 32x64 tile of C (2 M-tiles x 4 N-tiles); K marched in
// chunks of 32 with f16 WMMA, fp32 accumulation.  Fragment packing follows the
// CDNA5 ISA 16-bit A 16x32 layout and B 32x16 striping.
__global__ void __launch_bounds__(128, 1)
wmma_gemm_xwt(const float* __restrict__ A, int lda,
              const float* __restrict__ W, int ldw,
              const float* __restrict__ bias,
              float* __restrict__ C, int ldc,
              int M, int N, int K)
{
  const int lane = threadIdx.x & 31;
  const int wave = threadIdx.x >> 5;
  const int n0 = (blockIdx.x * 4 + wave) * 64;
  const int m0 = blockIdx.y * 32;
  const int half = lane >> 4;
  const int lm = lane & 15;

  // Clamp row pointers: out-of-range lanes read a valid row; their products
  // only pollute store-masked rows/cols of D.
  const float* arow[2];
  const float* wrow[4];
#pragma unroll
  for (int i = 0; i < 2; ++i)
    arow[i] = A + (size_t)min(m0 + 16 * i + lm, M - 1) * lda;
#pragma unroll
  for (int j = 0; j < 4; ++j)
    wrow[j] = W + (size_t)min(n0 + 16 * j + lm, N - 1) * ldw;

  v8f acc[2][4] = {};
  const int kFull = K & ~31;
  int kb = 0;
  for (; kb < kFull; kb += 32) {
    v16h a[2], b[4];
#pragma unroll
    for (int i = 0; i < 2; ++i) {
      const float* pa = arow[i] + kb + half * 8;   // two contiguous 8-f runs
#pragma unroll
      for (int u = 0; u < 8; ++u) {
        a[i][u]     = (_Float16)pa[u];             // A: VGPR0-3 K=half*8+0..7
        a[i][8 + u] = (_Float16)pa[16 + u];        //    VGPR4-7 K=16+half*8+..
      }
    }
#pragma unroll
    for (int j = 0; j < 4; ++j) {
      const float* pb = wrow[j] + kb + half * 8;   // two contiguous 8-f runs
#pragma unroll
      for (int u = 0; u < 16; ++u)
        b[j][u] = (_Float16)pb[u < 8 ? u : u + 8]; // B: same K order as A
    }
#pragma unroll
    for (int i = 0; i < 2; ++i)
#pragma unroll
      for (int j = 0; j < 4; ++j)
        acc[i][j] = __builtin_amdgcn_wmma_f32_16x16x32_f16(
            false, a[i], false, b[j], (short)0, acc[i][j], false, false);
  }
  if (kb < K) {                                    // branch-free K tail
    v16h a[2], b[4];
#pragma unroll
    for (int i = 0; i < 2; ++i) {
#pragma unroll
      for (int u = 0; u < 8; ++u) {
        int k0 = kb + half * 8 + u;
        int k1 = k0 + 16;
        float v0 = arow[i][min(k0, K - 1)];
        float v1 = arow[i][min(k1, K - 1)];
        a[i][u]     = (_Float16)(k0 < K ? v0 : 0.0f);
        a[i][8 + u] = (_Float16)(k1 < K ? v1 : 0.0f);
      }
    }
#pragma unroll
    for (int j = 0; j < 4; ++j) {
#pragma unroll
      for (int u = 0; u < 16; ++u) {
        int k = kb + half * 8 + (u < 8 ? u : u + 8);
        float v = wrow[j][min(k, K - 1)];
        b[j][u] = (_Float16)(k < K ? v : 0.0f);
      }
    }
#pragma unroll
    for (int i = 0; i < 2; ++i)
#pragma unroll
      for (int j = 0; j < 4; ++j)
        acc[i][j] = __builtin_amdgcn_wmma_f32_16x16x32_f16(
            false, a[i], false, b[j], (short)0, acc[i][j], false, false);
  }

#pragma unroll
  for (int j = 0; j < 4; ++j) {
    const int nc = n0 + 16 * j + lm;
    if (nc < N) {
      float bv = bias ? bias[nc] : 0.0f;
#pragma unroll
      for (int i = 0; i < 2; ++i) {
#pragma unroll
        for (int r = 0; r < 8; ++r) {              // C/D: VGPR r = row base+r
          int m = m0 + 16 * i + half * 8 + r;
          if (m < M) C[(size_t)m * ldc + nc] = acc[i][j][r] + bv;
        }
      }
    }
  }
}

// Generic single-layer LSTM recurrence.  grid = (2 dirs, up to 2 lstms of the
// same H).  blockDim = 4H.  Pre-gates G already contain Wi@x_t + b.
// Writes h into out[t][dir*H + k] (bilstm concat layout).
__global__ void lstm_rec(const float* __restrict__ Ga, const float* __restrict__ Gb,
                         const float* __restrict__ Wha, const float* __restrict__ Whb,
                         float* __restrict__ outa, float* __restrict__ outb,
                         int T, int H)
{
  const int dir = blockIdx.x;
  const int which = blockIdx.y;
  const float* G = (which ? Gb : Ga) + (size_t)dir * T * 4 * H;
  const float* Wh = (which ? Whb : Wha) + (size_t)dir * 4 * H * H;
  float* out = which ? outb : outa;

  extern __shared__ float sm[];
  float* hs = sm;                 // H
  float* gates = sm + H;          // 4H

  const int g = threadIdx.x;      // gate row 0..4H-1
  if (g < H) hs[g] = 0.0f;
  float c = 0.0f;
  __syncthreads();

  const float* wr = Wh + (size_t)g * H;
  for (int s = 0; s < T; ++s) {
    const int t = dir ? (T - 1 - s) : s;
    float acc = G[(size_t)t * 4 * H + g];
    for (int k = 0; k < H; ++k) acc += wr[k] * hs[k];
    gates[g] = acc;
    __syncthreads();
    if (g < H) {
      float ig = sigf(gates[g]);
      float fg = sigf(gates[H + g]);
      float gg = tanhf(gates[2 * H + g]);
      float og = sigf(gates[3 * H + g]);
      c = fg * c + ig * gg;
      float h = og * tanhf(c);
      hs[g] = h;
      out[(size_t)t * 2 * H + (size_t)dir * H + g] = h;
    }
    __syncthreads();
  }
}

// Feature-wise segment attention: segments are fixed runs of `seglen` rows
// (beat_ids = t/8, meas_of_beat = b/4).  grid = nseg, block = F.
__global__ void seg_attn(const float* __restrict__ logits, const float* __restrict__ h,
                         float* __restrict__ outn, int F, int seglen)
{
  const int sgm = blockIdx.x;
  const int f = threadIdx.x;
  float l[8];
  float m = -3.0e38f;
  for (int i = 0; i < seglen; ++i) {
    l[i] = logits[(size_t)(sgm * seglen + i) * F + f];
    m = fmaxf(m, l[i]);
  }
  float s = 0.0f, acc = 0.0f;
  for (int i = 0; i < seglen; ++i) {
    float e = expf(l[i] - m);
    s += e;
    acc += h[(size_t)(sgm * seglen + i) * F + f] * e;
  }
  outn[(size_t)sgm * F + f] = acc / s;
}

// Effective biases: fold bias + constant (qpm, tempo_primo) columns.
__global__ void prep_beff(const float* __restrict__ x,
                          const float* __restrict__ out_b, const float* __restrict__ outWi,
                          const float* __restrict__ bt_b,  const float* __restrict__ btWi,
                          const float* __restrict__ bfc,   const float* __restrict__ Wfc,
                          float* __restrict__ beffO, float* __restrict__ beffB,
                          float* __restrict__ beffF)
{
  const float qpm = x[5], tp0 = x[76], tp1 = x[77];
  const int g = threadIdx.x;
  beffO[g] = out_b[g] + outWi[(size_t)g * 1424 + 1421] * qpm
           + outWi[(size_t)g * 1424 + 1422] * tp0
           + outWi[(size_t)g * 1424 + 1423] * tp1;
  beffB[g] = bt_b[g] + btWi[(size_t)g * 273 + 257] * qpm
           + btWi[(size_t)g * 273 + 258] * tp0
           + btWi[(size_t)g * 273 + 259] * tp1;
  if (g < 10)
    beffF[g] = bfc[g] + Wfc[(size_t)g * 1552 + 1549] * qpm
             + Wfc[(size_t)g * 1552 + 1550] * tp0
             + Wfc[(size_t)g * 1552 + 1551] * tp1;
}

// Static part of comb: [note_h(512) | beat_h[t/8](256) | meas_h[t/32](128) | voice_h(512)]
__global__ void build_S(const float* __restrict__ noteH, const float* __restrict__ beatH,
                        const float* __restrict__ measH, const float* __restrict__ voiceH,
                        float* __restrict__ S)
{
  const int t = blockIdx.x;
  for (int c = threadIdx.x; c < 1408; c += blockDim.x) {
    float v;
    if (c < 512)      v = noteH[(size_t)t * 512 + c];
    else if (c < 768) v = beatH[(size_t)(t >> 3) * 256 + (c - 512)];
    else if (c < 896) v = measH[(size_t)(t >> 5) * 128 + (c - 768)];
    else              v = voiceH[(size_t)t * 512 + (c - 896)];
    S[(size_t)t * 1408 + c] = v;
  }
}

// Add bt_Wi's btv columns (x[:,28:31]) to the bt pre-gates.
__global__ void add_btv(float* __restrict__ Gbt, const float* __restrict__ btWi,
                        const float* __restrict__ x)
{
  const int t = blockIdx.x, g = threadIdx.x;
  const float* xv = x + (size_t)t * 78 + 28;
  const float* w = btWi + (size_t)g * 273 + 260;
  Gbt[(size_t)t * 512 + g] += w[0] * xv[0] + w[1] * xv[1] + w[2] * xv[2];
}

// The sequential output scan: out-LSTM + beat-tempo LSTM + buffer attention.
// Single 512-thread workgroup; all static projections precomputed.
__global__ void scan_kernel(const float* __restrict__ Gout, const float* __restrict__ Gbt,
                            const float* __restrict__ Fpre,
                            const float* __restrict__ outWi, const float* __restrict__ outWh,
                            const float* __restrict__ Wfc,
                            const float* __restrict__ btWi, const float* __restrict__ btWh,
                            const float* __restrict__ Wta, const float* __restrict__ bta,
                            const float* __restrict__ Wt1, const float* __restrict__ bt1,
                            const float* __restrict__ Wt2, const float* __restrict__ bt2,
                            const float* __restrict__ y, const int* __restrict__ bid,
                            float* __restrict__ outs, int T)
{
  __shared__ float oh[128], oc[128], th[128], tc[128], nh_[128], nc_[128], r1[128];
  __shared__ float gb[512], go[512];
  __shared__ float buf[80], rn[10], ofc[10], po[11], pv[2], ytail[10];
  __shared__ float ptv, tmpv;
  __shared__ int cnt, isb;

  const int tid = threadIdx.x;
  if (tid < 128) { oh[tid] = 0.f; oc[tid] = 0.f; th[tid] = 0.f; tc[tid] = 0.f; }
  if (tid < 80)  buf[tid] = 0.f;
  if (tid < 11)  po[tid] = y[tid];
  if (tid < 10)  ytail[tid] = y[1 + tid];
  if (tid < 2)   pv[tid] = 0.f;
  if (tid == 0) { ptv = y[0]; cnt = 0; }
  __syncthreads();

  for (int t = 0; t < T; ++t) {
    // --- result_node: feature-wise attention over the note buffer ----------
    if (tid < 10) {
      const int c = cnt;
      if (c == 0) rn[tid] = ytail[tid];
      else {
        float l[8]; float m = -3.0e38f;
        for (int r = 0; r < c; ++r) {
          float a = bta[tid];
          for (int j = 0; j < 10; ++j) a += buf[r * 10 + j] * Wta[tid * 10 + j];
          l[r] = a; m = fmaxf(m, a);
        }
        float s = 0.f, acc = 0.f;
        for (int r = 0; r < c; ++r) {
          float e = expf(l[r] - m);
          s += e; acc += buf[r * 10 + tid] * e;
        }
        rn[tid] = acc / s;
      }
    }
    if (tid == 0) isb = (t == 0) || (bid[t] > bid[t - 1]);
    __syncthreads();
    // --- beat-tempo LSTM gates (recurrent part only) -----------------------
    {
      const float* wi = btWi + (size_t)tid * 273;
      const float* wh = btWh + (size_t)tid * 128;
      float acc = Gbt[(size_t)t * 512 + tid] + wi[256] * ptv;
      for (int j = 0; j < 10; ++j) acc += wi[263 + j] * rn[j];
      for (int k = 0; k < 128; ++k) acc += wh[k] * th[k];
      gb[tid] = acc;
    }
    __syncthreads();
    if (tid < 128) {
      float ig = sigf(gb[tid]), fg = sigf(gb[128 + tid]);
      float gg = tanhf(gb[256 + tid]), og = sigf(gb[384 + tid]);
      float c2 = fg * tc[tid] + ig * gg;
      nc_[tid] = c2; nh_[tid] = og * tanhf(c2);
    }
    __syncthreads();
    if (tid < 128) {
      float a = bt1[tid];
      const float* w = Wt1 + (size_t)tid * 128;
      for (int j = 0; j < 128; ++j) a += w[j] * nh_[j];
      r1[tid] = fmaxf(a, 0.f);
    }
    __syncthreads();
    if (tid == 0) {
      float a = bt2[0];
      for (int j = 0; j < 128; ++j) a += Wt2[j] * r1[j];
      tmpv = a;
    }
    __syncthreads();
    if (isb) {                       // commit beat-tempo state at beat starts
      if (tid < 128) { th[tid] = nh_[tid]; tc[tid] = nc_[tid]; }
      if (tid < 80)  buf[tid] = 0.f;
      if (tid == 0)  { ptv = tmpv; cnt = 0; }
    }
    __syncthreads();
    // --- output LSTM gates (recurrent part only) ---------------------------
    {
      const float* wi = outWi + (size_t)tid * 1424;
      const float* wh = outWh + (size_t)tid * 128;
      float acc = Gout[(size_t)t * 512 + tid];
      for (int j = 0; j < 11; ++j) acc += wi[1408 + j] * po[j];
      acc += wi[1419] * pv[0] + wi[1420] * pv[1];
      for (int k = 0; k < 128; ++k) acc += wh[k] * oh[k];
      go[tid] = acc;
    }
    __syncthreads();
    if (tid < 128) {
      float ig = sigf(go[tid]), fg = sigf(go[128 + tid]);
      float gg = tanhf(go[256 + tid]), og = sigf(go[384 + tid]);
      float c2 = fg * oc[tid] + ig * gg;
      oc[tid] = c2; oh[tid] = og * tanhf(c2);
    }
    __syncthreads();
    // --- final FC (recurrent part only) ------------------------------------
    if (tid < 10) {
      const float* w = Wfc + (size_t)tid * 1552;
      float a = Fpre[(size_t)t * 10 + tid];
      for (int k = 0; k < 128; ++k) a += w[k] * oh[k];
      for (int j = 0; j < 11; ++j) a += w[1536 + j] * po[j];
      a += w[1547] * pv[0] + w[1548] * pv[1];
      ofc[tid] = a;
    }
    __syncthreads();
    // --- commit outputs / carries ------------------------------------------
    if (tid == 0) { outs[(size_t)t * 11] = ptv; po[0] = ptv; }
    if (tid < 10) {
      outs[(size_t)t * 11 + 1 + tid] = ofc[tid];
      po[1 + tid] = ofc[tid];
      buf[cnt * 10 + tid] = ofc[tid];
    }
    if (tid < 2) pv[tid] = ofc[tid];
    __syncthreads();
    if (tid == 0) cnt = cnt + 1;
    __syncthreads();
  }
}

// ---------------------------------------------------------------------------
extern "C" void kernel_launch(void* const* d_in, const int* in_sizes, int n_in,
                              void* d_out, int out_size, void* d_ws, size_t ws_size,
                              hipStream_t stream) {
  (void)in_sizes; (void)n_in; (void)out_size; (void)ws_size;
  const float* x       = (const float*)d_in[0];
  const float* y       = (const float*)d_in[1];
  const int*   bid     = (const int*)d_in[2];
  const float* noteWi  = (const float*)d_in[4];
  const float* noteWh  = (const float*)d_in[5];
  const float* note_b  = (const float*)d_in[6];
  const float* voiceWi = (const float*)d_in[7];
  const float* voiceWh = (const float*)d_in[8];
  const float* voice_b = (const float*)d_in[9];
  const float* beatWi  = (const float*)d_in[10];
  const float* beatWh  = (const float*)d_in[11];
  const float* beat_b  = (const float*)d_in[12];
  const float* measWi  = (const float*)d_in[13];
  const float* measWh  = (const float*)d_in[14];
  const float* meas_b  = (const float*)d_in[15];
  const float* Wba     = (const float*)d_in[16];
  const float* bba     = (const float*)d_in[17];
  const float* Wma     = (const float*)d_in[18];
  const float* bma     = (const float*)d_in[19];
  const float* Wta     = (const float*)d_in[20];
  const float* bta     = (const float*)d_in[21];
  const float* outWi   = (const float*)d_in[22];
  const float* outWh   = (const float*)d_in[23];
  const float* out_b   = (const float*)d_in[24];
  const float* Wfc     = (const float*)d_in[25];
  const float* bfc     = (const float*)d_in[26];
  const float* btWi    = (const float*)d_in[27];
  const float* btWh    = (const float*)d_in[28];
  const float* bt_b    = (const float*)d_in[29];
  const float* Wt1     = (const float*)d_in[30];
  const float* bt1     = (const float*)d_in[31];
  const float* Wt2     = (const float*)d_in[32];
  const float* bt2     = (const float*)d_in[33];

  float* outF  = (float*)d_out;                 // outs: 4096*11
  float* noteH = (float*)d_out + 45056;         // note_h: 4096*512 (2nd output)
  float* ws    = (float*)d_ws;

  // Workspace layout (floats).  GN/GV are dead after the note/voice
  // recurrences and are reused for S/Gout/Gbt/Fpre.
  const size_t GN = 0,                 GV = 8388608;
  const size_t S_ = 0,                 GOUT = 5767168, GBT = 7864320, FPRE = 9961472;
  const size_t VOICEH = 16777216,      LBA = 18874368, BNODES = 20971520;
  const size_t GB_ = 21233664,         BEATH = 21757952, LMA = 21889024;
  const size_t MNODES = 22020096,      GM_ = 22052864, MEASH = 22118400;
  const size_t BEFFO = 22134784,       BEFFB = 22135296, BEFFF = 22135808;

  auto gemm = [&](const float* A, int lda, const float* W, int ldw,
                  const float* bias, float* C, int ldc, int M, int N, int K) {
    dim3 grid((N + 255) / 256, (M + 31) / 32);
    wmma_gemm_xwt<<<grid, 128, 0, stream>>>(A, lda, W, ldw, bias, C, ldc, M, N, K);
  };

  prep_beff<<<1, 512, 0, stream>>>(x, out_b, outWi, bt_b, btWi, bfc, Wfc,
                                   ws + BEFFO, ws + BEFFB, ws + BEFFF);

  // Input projections for note & voice BiLSTMs (WMMA).
  gemm(x, 78, noteWi,               78, note_b,        ws + GN,               1024, 4096, 1024, 78);
  gemm(x, 78, noteWi + 1024 * 78,   78, note_b + 1024, ws + GN + 4096 * 1024, 1024, 4096, 1024, 78);
  gemm(x, 78, voiceWi,              78, voice_b,       ws + GV,               1024, 4096, 1024, 78);
  gemm(x, 78, voiceWi + 1024 * 78,  78, voice_b + 1024,ws + GV + 4096 * 1024, 1024, 4096, 1024, 78);

  // note & voice recurrences: 4 concurrent chains (2 lstms x 2 dirs).
  lstm_rec<<<dim3(2, 2), 1024, 5 * 256 * 4, stream>>>(
      ws + GN, ws + GV, noteWh, voiceWh, noteH, ws + VOICEH, 4096, 256);

  // Beat attention + beat BiLSTM.
  gemm(noteH, 512, Wba, 512, bba, ws + LBA, 512, 4096, 512, 512);
  seg_attn<<<512, 512, 0, stream>>>(ws + LBA, noteH, ws + BNODES, 512, 8);
  gemm(ws + BNODES, 512, beatWi,             512, beat_b,       ws + GB_,             512, 512, 512, 512);
  gemm(ws + BNODES, 512, beatWi + 512 * 512, 512, beat_b + 512, ws + GB_ + 512 * 512, 512, 512, 512, 512);
  lstm_rec<<<dim3(2, 1), 512, 5 * 128 * 4, stream>>>(
      ws + GB_, ws + GB_, beatWh, beatWh, ws + BEATH, ws + BEATH, 512, 128);

  // Measure attention + measure BiLSTM.
  gemm(ws + BEATH, 256, Wma, 256, bma, ws + LMA, 256, 512, 256, 256);
  seg_attn<<<128, 256, 0, stream>>>(ws + LMA, ws + BEATH, ws + MNODES, 256, 4);
  gemm(ws + MNODES, 256, measWi,             256, meas_b,       ws + GM_,             256, 128, 256, 256);
  gemm(ws + MNODES, 256, measWi + 256 * 256, 256, meas_b + 256, ws + GM_ + 128 * 256, 256, 128, 256, 256);
  lstm_rec<<<dim3(2, 1), 256, 5 * 64 * 4, stream>>>(
      ws + GM_, ws + GM_, measWh, measWh, ws + MEASH, ws + MEASH, 128, 64);

  // Static comb matrix + big precomputed projections for the scan (WMMA).
  build_S<<<4096, 256, 0, stream>>>(noteH, ws + BEATH, ws + MEASH, ws + VOICEH, ws + S_);
  gemm(ws + S_,       1408, outWi,      1424, ws + BEFFO, ws + GOUT, 512, 4096, 512, 1408);
  gemm(ws + S_ + 512, 1408, btWi,       273,  ws + BEFFB, ws + GBT,  512, 4096, 512, 256);
  add_btv<<<4096, 512, 0, stream>>>(ws + GBT, btWi, x);
  gemm(ws + S_,       1408, Wfc + 128,  1552, ws + BEFFF, ws + FPRE, 10,  4096, 10,  1408);

  // Sequential output scan.
  scan_kernel<<<1, 512, 0, stream>>>(ws + GOUT, ws + GBT, ws + FPRE,
                                     outWi, outWh, Wfc, btWi, btWh,
                                     Wta, bta, Wt1, bt1, Wt2, bt2,
                                     y, bid, outF, 4096);
}
